// AdaptiveSoftmax_36249523978666
// MI455X (gfx1250) — compile-verified, matrix operand order checked by reading, not run on hardware
//
#include <hip/hip_runtime.h>
#include <hip/hip_bf16.h>
#include <math.h>

typedef __attribute__((ext_vector_type(16))) __bf16 v16bf;
typedef __attribute__((ext_vector_type(8)))  __bf16 v8bf;
typedef __attribute__((ext_vector_type(8)))  float  v8f;

union V16 { v16bf v; v8bf h[2]; };

#define INPUT_DIM 1024

// ---------------- conversion kernels ----------------

__global__ void k_cvt_bf16(const float* __restrict__ src, __bf16* __restrict__ dst, int n) {
    int i = blockIdx.x * blockDim.x + threadIdx.x;
    if (i < n) dst[i] = (__bf16)src[i];
}

// dst[c*K + k] = (c < C) ? src[k*C + c] : 0   ; dst has Cp rows (Cp >= C, padded)
__global__ void k_cvt_tr(const float* __restrict__ src, __bf16* __restrict__ dst,
                         int K, int C, int Cp) {
    int i = blockIdx.x * blockDim.x + threadIdx.x;
    int total = Cp * K;
    if (i >= total) return;
    int c = i / K;
    int k = i - c * K;
    dst[i] = (c < C) ? (__bf16)src[k * C + c] : (__bf16)0.0f;
}

// ---------------- GEMM1: Hg = gelu(Xb @ P) ----------------
// Xb: [N x 1024] bf16 row-major.  Pt: [Cout x 1024] bf16 (P transposed).
// block = 128 threads (4 waves); wave computes one 16x16 tile. grid = (N/16, Cout/64)
__global__ __launch_bounds__(128) void k_gemm1_gelu(
    const __bf16* __restrict__ Xb, const __bf16* __restrict__ Pt,
    __bf16* __restrict__ Hg, int Cout) {
    const int K = INPUT_DIM;
    int lane = threadIdx.x & 31;
    int wave = threadIdx.x >> 5;
    int m = blockIdx.x;
    int c0 = blockIdx.y * 64 + wave * 16;
    int r    = lane & 15;
    int koff = (lane >> 4) * 8;

    const __bf16* arow = Xb + (long long)(m * 16 + r) * K;
    const __bf16* brow = Pt + (long long)(c0 + r) * K;

    v8f acc = {};
    for (int kb = 0; kb < K; kb += 32) {
        V16 a, b;
        a.h[0] = *reinterpret_cast<const v8bf*>(arow + kb + koff);
        a.h[1] = *reinterpret_cast<const v8bf*>(arow + kb + 16 + koff);
        b.h[0] = *reinterpret_cast<const v8bf*>(brow + kb + koff);
        b.h[1] = *reinterpret_cast<const v8bf*>(brow + kb + 16 + koff);
        acc = __builtin_amdgcn_wmma_f32_16x16x32_bf16(
            false, a.v, false, b.v, (short)0, acc, false, false);
    }

    int col   = c0 + r;
    int rbase = (lane >> 4) * 8;
    #pragma unroll
    for (int v = 0; v < 8; v++) {
        float x = acc[v];
        float g = x * 0.5f * (1.0f + erff(x * 0.70710678118654752f));
        Hg[(long long)(m * 16 + rbase + v) * Cout + col] = (__bf16)g;
    }
}

// ---------------- GEMM2 + fused online-softmax cross-entropy ----------------
// Block = 256 threads (8 waves) owns a 32-row stripe. A stripe staged in LDS via
// async global->LDS (+8-elem row pad => conflict-free ds_load_b128). Each wave
// computes a 32x16 tile per class tile (2 accumulators, B reused 2x), striding
// over class tiles. Online logsumexp per lane, label logit captured to LDS.
template<int KP>
__global__ __launch_bounds__(256) void k_gemm2_ce(
    const __bf16* __restrict__ Hg, const __bf16* __restrict__ Wt,
    const float* __restrict__ bias, const int* __restrict__ labels,
    float* __restrict__ ce_out, int C, int Cpad, int mode, int lo) {
    constexpr int KPAD = KP + 8;               // bf16 elems per padded LDS row
    extern __shared__ __bf16 sA[];             // [32 * KPAD]
    __shared__ int   s_lab[32];
    __shared__ float s_lablogit[32];
    __shared__ float s_m[8][32];
    __shared__ float s_s[8][32];

    const int tid  = threadIdx.x;
    const int lane = tid & 31;
    const int wave = tid >> 5;
    const int m    = blockIdx.x;               // 32-row stripe index

    if (tid < 32) {
        int lab = labels[m * 32 + tid];
        if (mode == 0) {
            if (lab >= 10000)     lab = 2001;
            else if (lab >= 2000) lab = 2000;
        } else {
            lab -= lo;
            lab = lab < 0 ? 0 : (lab > C - 1 ? C - 1 : lab);
        }
        s_lab[tid] = lab;
        s_lablogit[tid] = 0.0f;
    }

    // ---- stage A stripe (32 x KP bf16) into padded LDS via async copy ----
    {
        constexpr int CH = KP / 8;             // 16B chunks per row
        for (int idx = tid; idx < 32 * CH; idx += 256) {
            int row = idx / CH;
            int kc  = (idx - row * CH) * 8;
            const __bf16* g = Hg + (long long)(m * 32 + row) * KP + kc;
            unsigned ldsb = (unsigned)(size_t)(sA + row * KPAD + kc);
            asm volatile("global_load_async_to_lds_b128 %0, %1, off"
                         :: "v"(ldsb), "v"(g) : "memory");
        }
        asm volatile("s_wait_asynccnt 0" ::: "memory");
    }
    __syncthreads();

    const int r     = lane & 15;
    const int koff  = (lane >> 4) * 8;
    const int rbase = (lane >> 4) * 8;
    const __bf16* a0 = sA + r * KPAD;          // rows 0-15 of stripe
    const __bf16* a1 = sA + (16 + r) * KPAD;   // rows 16-31

    float rm[2][8], rs[2][8];
    #pragma unroll
    for (int rt = 0; rt < 2; rt++)
        #pragma unroll
        for (int v = 0; v < 8; v++) { rm[rt][v] = -INFINITY; rs[rt][v] = 0.0f; }

    const int numTiles = Cpad >> 4;
    for (int t = wave; t < numTiles; t += 8) {
        const int col = t * 16 + r;
        const __bf16* brow = Wt + (long long)col * KP;
        if (t + 8 < numTiles)
            __builtin_prefetch(Wt + (long long)(col + 128) * KP, 0, 1);

        v8f acc0 = {}, acc1 = {};
        #pragma unroll
        for (int kb = 0; kb < KP; kb += 32) {
            V16 b, x0, x1;
            b.h[0]  = *reinterpret_cast<const v8bf*>(brow + kb + koff);
            b.h[1]  = *reinterpret_cast<const v8bf*>(brow + kb + 16 + koff);
            x0.h[0] = *reinterpret_cast<const v8bf*>(a0 + kb + koff);
            x0.h[1] = *reinterpret_cast<const v8bf*>(a0 + kb + 16 + koff);
            x1.h[0] = *reinterpret_cast<const v8bf*>(a1 + kb + koff);
            x1.h[1] = *reinterpret_cast<const v8bf*>(a1 + kb + 16 + koff);
            acc0 = __builtin_amdgcn_wmma_f32_16x16x32_bf16(
                false, x0.v, false, b.v, (short)0, acc0, false, false);
            acc1 = __builtin_amdgcn_wmma_f32_16x16x32_bf16(
                false, x1.v, false, b.v, (short)0, acc1, false, false);
        }

        if (col < C) {
            float bv = bias[col];
            #pragma unroll
            for (int rt = 0; rt < 2; rt++) {
                #pragma unroll
                for (int v = 0; v < 8; v++) {
                    float x = (rt ? acc1[v] : acc0[v]) + bv;
                    int row = rt * 16 + rbase + v;
                    if (s_lab[row] == col) s_lablogit[row] = x;  // one writer/row
                    if (x > rm[rt][v]) {                          // rescale on new max
                        rs[rt][v] *= __expf(rm[rt][v] - x);
                        rm[rt][v] = x;
                    }
                    rs[rt][v] += __expf(x - rm[rt][v]);
                }
            }
        }
    }

    // merge logsumexp state across the 16 columns within each lane-half
    #pragma unroll
    for (int d = 1; d < 16; d <<= 1) {
        #pragma unroll
        for (int rt = 0; rt < 2; rt++) {
            #pragma unroll
            for (int v = 0; v < 8; v++) {
                float om = __shfl_xor(rm[rt][v], d, 32);
                float os = __shfl_xor(rs[rt][v], d, 32);
                float M = fmaxf(rm[rt][v], om);
                float s = (M == -INFINITY) ? 0.0f
                        : rs[rt][v] * __expf(rm[rt][v] - M) + os * __expf(om - M);
                rm[rt][v] = M; rs[rt][v] = s;
            }
        }
    }
    if ((lane & 15) == 0) {                    // lanes 0 and 16
        #pragma unroll
        for (int rt = 0; rt < 2; rt++)
            #pragma unroll
            for (int v = 0; v < 8; v++) {
                s_m[wave][rt * 16 + rbase + v] = rm[rt][v];
                s_s[wave][rt * 16 + rbase + v] = rs[rt][v];
            }
    }
    __syncthreads();

    if (tid < 32) {
        float M = -INFINITY, S = 0.0f;
        #pragma unroll
        for (int w = 0; w < 8; w++) {
            float om = s_m[w][tid], os = s_s[w][tid];
            float nm = fmaxf(M, om);
            if (nm != -INFINITY)
                S = S * __expf(M - nm) + os * __expf(om - nm);
            M = nm;
        }
        float lse = M + __logf(S);
        ce_out[m * 32 + tid] = lse - s_lablogit[tid];
    }
}

// ---------------- final masked combine ----------------
__global__ void k_final(const float* __restrict__ ceH, const float* __restrict__ ce1,
                        const float* __restrict__ ce2, const int* __restrict__ labels,
                        float* __restrict__ out, int n) {
    int i = blockIdx.x * blockDim.x + threadIdx.x;
    if (i >= n) return;
    int lab = labels[i];
    float l = ceH[i];
    if (lab >= 2000 && lab < 10000)       l += ce1[i];
    else if (lab >= 10000 && lab < 50000) l += ce2[i];
    out[i] = l;
}

// ---------------- launcher ----------------
extern "C" void kernel_launch(void* const* d_in, const int* in_sizes, int n_in,
                              void* d_out, int out_size, void* d_ws, size_t ws_size,
                              hipStream_t stream) {
    const float* x          = (const float*)d_in[0];
    const int*   labels     = (const int*)  d_in[1];
    const float* head_proj  = (const float*)d_in[2];
    const float* head_w     = (const float*)d_in[3];
    const float* head_b     = (const float*)d_in[4];
    const float* t1_proj    = (const float*)d_in[5];
    const float* t1_w       = (const float*)d_in[6];
    const float* t1_b       = (const float*)d_in[7];
    const float* t2_proj    = (const float*)d_in[8];
    const float* t2_w       = (const float*)d_in[9];
    const float* t2_b       = (const float*)d_in[10];
    float* out = (float*)d_out;

    const int N  = in_sizes[1];      // 4096
    const int K  = INPUT_DIM;        // 1024
    const int CH = 2002, CHp = 2016;
    const int C1 = 8000,  Kp1 = 256;
    const int C2 = 40000, Kp2 = 64;

    char* ws = (char*)d_ws;
    size_t off = 0;
    auto alloc = [&](size_t bytes) { void* p = ws + off; off += (bytes + 255) & ~(size_t)255; return p; };
    __bf16* Xb  = (__bf16*)alloc((size_t)N * K * 2);
    __bf16* PtH = (__bf16*)alloc((size_t)K * K * 2);
    __bf16* Pt1 = (__bf16*)alloc((size_t)Kp1 * K * 2);
    __bf16* Pt2 = (__bf16*)alloc((size_t)Kp2 * K * 2);
    __bf16* WtH = (__bf16*)alloc((size_t)CHp * K * 2);
    __bf16* Wt1 = (__bf16*)alloc((size_t)C1 * Kp1 * 2);
    __bf16* Wt2 = (__bf16*)alloc((size_t)C2 * Kp2 * 2);
    __bf16* HgH = (__bf16*)alloc((size_t)N * K * 2);
    __bf16* Hg1 = (__bf16*)alloc((size_t)N * Kp1 * 2);
    __bf16* Hg2 = (__bf16*)alloc((size_t)N * Kp2 * 2);
    float*  ceH = (float*)alloc((size_t)N * 4);
    float*  ce1 = (float*)alloc((size_t)N * 4);
    float*  ce2 = (float*)alloc((size_t)N * 4);
    (void)ws_size; (void)n_in; (void)out_size;

    {
        int n = N * K;
        k_cvt_bf16<<<(n + 255) / 256, 256, 0, stream>>>(x, Xb, n);
    }
    auto tr = [&](const float* src, __bf16* dst, int k, int c, int cp) {
        int total = cp * k;
        k_cvt_tr<<<(total + 255) / 256, 256, 0, stream>>>(src, dst, k, c, cp);
    };
    tr(head_proj, PtH, K, K,   K);
    tr(t1_proj,   Pt1, K, Kp1, Kp1);
    tr(t2_proj,   Pt2, K, Kp2, Kp2);
    tr(head_w,    WtH, K,   CH, CHp);
    tr(t1_w,      Wt1, Kp1, C1, C1);
    tr(t2_w,      Wt2, Kp2, C2, C2);

    k_gemm1_gelu<<<dim3(N / 16, K   / 64), 128, 0, stream>>>(Xb, PtH, HgH, K);
    k_gemm1_gelu<<<dim3(N / 16, Kp1 / 64), 128, 0, stream>>>(Xb, Pt1, Hg1, Kp1);
    k_gemm1_gelu<<<dim3(N / 16, Kp2 / 64), 128, 0, stream>>>(Xb, Pt2, Hg2, Kp2);

    k_gemm2_ce<1024><<<N / 32, 256, 32 * (1024 + 8) * 2, stream>>>(
        HgH, WtH, head_b, labels, ceH, CH, CHp, 0, 0);
    k_gemm2_ce<256><<<N / 32, 256, 32 * (256 + 8) * 2, stream>>>(
        Hg1, Wt1, t1_b, labels, ce1, C1, C1, 1, 2000);
    k_gemm2_ce<64><<<N / 32, 256, 32 * (64 + 8) * 2, stream>>>(
        Hg2, Wt2, t2_b, labels, ce2, C2, C2, 1, 10000);

    k_final<<<(N + 255) / 256, 256, 0, stream>>>(ceH, ce1, ce2, labels, out, N);
}
